// GraphConvBlock_39822936768632
// MI455X (gfx1250) — matrix operand order repaired and checked by reading, not compiled, not run on hardware
//
#include <hip/hip_runtime.h>
#include <hip/hip_bf16.h>

#define DFEAT 512
#define LN_EPS 1e-5f

typedef __attribute__((ext_vector_type(16))) __bf16 v16bf;
typedef __attribute__((ext_vector_type(8)))  float  v8f;

// round-to-nearest-even f32 -> bf16, packed pair (lo = first element)
__device__ __forceinline__ unsigned int pack_bf16(float a, float b) {
    unsigned int ua = __float_as_uint(a);
    unsigned int ub = __float_as_uint(b);
    ua += 0x7FFFu + ((ua >> 16) & 1u);
    ub += 0x7FFFu + ((ub >> 16) & 1u);
    return (ua >> 16) | (ub & 0xFFFF0000u);
}

// ---------------- utility: zero fill (graph-capture safe) ----------------
__global__ void zero_f4(float4* __restrict__ p, long n4) {
    long i = (long)blockIdx.x * blockDim.x + threadIdx.x;
    long stride = (long)gridDim.x * blockDim.x;
    float4 z = {0.f, 0.f, 0.f, 0.f};
    for (; i < n4; i += stride) p[i] = z;
}

// ---------------- f32 -> bf16 bulk convert (used for W and for agg) ----------------
__global__ void bf16conv_kernel(const float4* __restrict__ in, uint2* __restrict__ outb, long n4) {
    long i = (long)blockIdx.x * blockDim.x + threadIdx.x;
    long stride = (long)gridDim.x * blockDim.x;
    for (; i < n4; i += stride) {
        float4 f = in[i];
        uint2 o;
        o.x = pack_bf16(f.x, f.y);
        o.y = pack_bf16(f.z, f.w);
        outb[i] = o;
    }
}

// ---------------- degree count ----------------
__global__ void deg_kernel(const int* __restrict__ dst, float* __restrict__ deg, int E) {
    int e = blockIdx.x * blockDim.x + threadIdx.x;
    if (e < E) atomicAdd(&deg[dst[e]], 1.0f);
}

// ---------------- norm = clip(deg,1)^-0.5 ----------------
__global__ void norm_kernel(const float* __restrict__ deg, float* __restrict__ norm, int N) {
    int n = blockIdx.x * blockDim.x + threadIdx.x;
    if (n < N) norm[n] = rsqrtf(fmaxf(deg[n], 1.0f));
}

// ---------------- edge scatter: agg[dst] += feat[src]*w*norm[src]*norm[dst] ----------------
// one wave32 per edge; feat/agg are L2-resident (102MB each < 192MB L2)
__global__ __launch_bounds__(256) void scatter_kernel(
    const float* __restrict__ feat, const int* __restrict__ src, const int* __restrict__ dst,
    const float* __restrict__ w, const float* __restrict__ norm, float* __restrict__ agg, int E)
{
    int e = blockIdx.x * 8 + (threadIdx.x >> 5);
    if (e >= E) return;
    int lane = threadIdx.x & 31;
    int s = src[e], d = dst[e];
    float coef = w[e] * norm[s] * norm[d];
    const float4* fp = (const float4*)(feat + (size_t)s * DFEAT);
    float* ap = agg + (size_t)d * DFEAT;
#pragma unroll
    for (int i = 0; i < 4; ++i) {
        float4 v = fp[lane + i * 32];
        int c = (lane + i * 32) * 4;
        atomicAdd(ap + c + 0, v.x * coef);
        atomicAdd(ap + c + 1, v.y * coef);
        atomicAdd(ap + c + 2, v.z * coef);
        atomicAdd(ap + c + 3, v.w * coef);
    }
}

// ---------------- fused GEMM (bf16 WMMA) + bias + LayerNorm + residual + ReLU ----
// block = 256 thr (8 waves) computes rows [m0, m0+32) x all 512 cols.
// wave w: 32x64 tile = 2 A-fragments x 4 B-tiles = 8 accumulators; each B fragment
// feeds two WMMAs (halves W L2 traffic vs 16-row tiles). K-loop 16 x 32.
__global__ __launch_bounds__(256) void gemm_ln_kernel(
    const unsigned short* __restrict__ aggb, const unsigned short* __restrict__ Wb,
    const float* __restrict__ bias, const float* __restrict__ gamma,
    const float* __restrict__ beta, const float* __restrict__ feat,
    float* __restrict__ out, int N)
{
    __shared__ float lds[32 * 516];   // 32 rows x 512 cols, padded stride
    const int tid   = threadIdx.x;
    const int wave  = tid >> 5;
    const int lane  = tid & 31;
    const int l15   = lane & 15;
    const int lhalf = lane >> 4;      // 0 | 1
    const int m0 = blockIdx.x * 32;
    const int j0 = wave * 64;

    v8f p0 = {}, p1 = {}, p2 = {}, p3 = {};   // rows m0..m0+15
    v8f q0 = {}, q1 = {}, q2 = {}, q3 = {};   // rows m0+16..m0+31

    // A 16x32 bf16: lane row; elems 0-15 = K kb..kb+15,
    // kb = k0 + 16*lhalf. Two b128 loads per fragment per K-step. Clamp rows for the tail block.
    int r0 = m0 + l15;      if (r0 > N - 1) r0 = N - 1;
    int r1 = m0 + 16 + l15; if (r1 > N - 1) r1 = N - 1;
    const unsigned short* ap0 = aggb + (size_t)r0 * DFEAT + lhalf * 16;
    const unsigned short* ap1 = aggb + (size_t)r1 * DFEAT + lhalf * 16;
    // B 32x16 bf16: lane = output column; lanes 0-15 hold K k0..k0+15, lanes 16-31 hold K k0+16..k0+31.
    const unsigned short* wp0 = Wb + (size_t)(j0 +  0 + l15) * DFEAT + lhalf * 16;
    const unsigned short* wp1 = Wb + (size_t)(j0 + 16 + l15) * DFEAT + lhalf * 16;
    const unsigned short* wp2 = Wb + (size_t)(j0 + 32 + l15) * DFEAT + lhalf * 16;
    const unsigned short* wp3 = Wb + (size_t)(j0 + 48 + l15) * DFEAT + lhalf * 16;

    for (int k0 = 0; k0 < DFEAT; k0 += 32) {
        union { v16bf v; uint4 q[2]; } A0, A1, B0, B1, B2, B3;
        A0.q[0] = *(const uint4*)(ap0 + k0);
        A0.q[1] = *(const uint4*)(ap0 + k0 + 8);
        A1.q[0] = *(const uint4*)(ap1 + k0);
        A1.q[1] = *(const uint4*)(ap1 + k0 + 8);
        B0.q[0] = *(const uint4*)(wp0 + k0);
        B0.q[1] = *(const uint4*)(wp0 + k0 + 8);
        B1.q[0] = *(const uint4*)(wp1 + k0);
        B1.q[1] = *(const uint4*)(wp1 + k0 + 8);
        B2.q[0] = *(const uint4*)(wp2 + k0);
        B2.q[1] = *(const uint4*)(wp2 + k0 + 8);
        B3.q[0] = *(const uint4*)(wp3 + k0);
        B3.q[1] = *(const uint4*)(wp3 + k0 + 8);

        p0 = __builtin_amdgcn_wmma_f32_16x16x32_bf16(false, A0.v, false, B0.v, (short)0, p0, false, false);
        q0 = __builtin_amdgcn_wmma_f32_16x16x32_bf16(false, A1.v, false, B0.v, (short)0, q0, false, false);
        p1 = __builtin_amdgcn_wmma_f32_16x16x32_bf16(false, A0.v, false, B1.v, (short)0, p1, false, false);
        q1 = __builtin_amdgcn_wmma_f32_16x16x32_bf16(false, A1.v, false, B1.v, (short)0, q1, false, false);
        p2 = __builtin_amdgcn_wmma_f32_16x16x32_bf16(false, A0.v, false, B2.v, (short)0, p2, false, false);
        q2 = __builtin_amdgcn_wmma_f32_16x16x32_bf16(false, A1.v, false, B2.v, (short)0, q2, false, false);
        p3 = __builtin_amdgcn_wmma_f32_16x16x32_bf16(false, A0.v, false, B3.v, (short)0, p3, false, false);
        q3 = __builtin_amdgcn_wmma_f32_16x16x32_bf16(false, A1.v, false, B3.v, (short)0, q3, false, false);
    }

    // C/D layout: lane = col (mod 16), VGPR v = row rb+v, rb = 8*lhalf. Park (+bias) in LDS.
    const int rb = lhalf * 8;
    const float b0 = bias[j0 +  0 + l15];
    const float b1 = bias[j0 + 16 + l15];
    const float b2 = bias[j0 + 32 + l15];
    const float b3 = bias[j0 + 48 + l15];
#pragma unroll
    for (int v = 0; v < 8; ++v) {
        int ra = rb + v;
        int rc = 16 + rb + v;
        lds[ra * 516 + j0 +  0 + l15] = p0[v] + b0;
        lds[ra * 516 + j0 + 16 + l15] = p1[v] + b1;
        lds[ra * 516 + j0 + 32 + l15] = p2[v] + b2;
        lds[ra * 516 + j0 + 48 + l15] = p3[v] + b3;
        lds[rc * 516 + j0 +  0 + l15] = q0[v] + b0;
        lds[rc * 516 + j0 + 16 + l15] = q1[v] + b1;
        lds[rc * 516 + j0 + 32 + l15] = q2[v] + b2;
        lds[rc * 516 + j0 + 48 + l15] = q3[v] + b3;
    }
    __syncthreads();

    // LayerNorm + residual + ReLU: each wave handles 4 full rows via wave32 shuffle reduction
#pragma unroll
    for (int rr = 0; rr < 4; ++rr) {
        int r = wave * 4 + rr;
        int grow = m0 + r;
        if (grow >= N) continue;     // uniform per wave: whole wave skips together
        float vals[16];
        float s = 0.f, s2 = 0.f;
#pragma unroll
        for (int i = 0; i < 16; ++i) {
            float x = lds[r * 516 + lane + 32 * i];
            vals[i] = x; s += x; s2 += x * x;
        }
#pragma unroll
        for (int off = 16; off > 0; off >>= 1) {
            s  += __shfl_xor(s,  off, 32);
            s2 += __shfl_xor(s2, off, 32);
        }
        float mu  = s  * (1.0f / 512.0f);
        float var = s2 * (1.0f / 512.0f) - mu * mu;
        float rs  = rsqrtf(var + LN_EPS);
        const float* frow = feat + (size_t)grow * DFEAT;
        float* orow = out + (size_t)grow * DFEAT;
#pragma unroll
        for (int i = 0; i < 16; ++i) {
            int c = lane + 32 * i;
            float y = (vals[i] - mu) * rs * gamma[c] + beta[c] + frow[c];
            orow[c] = fmaxf(y, 0.0f);
        }
    }
}

extern "C" void kernel_launch(void* const* d_in, const int* in_sizes, int n_in,
                              void* d_out, int out_size, void* d_ws, size_t ws_size,
                              hipStream_t stream)
{
    const float* feat  = (const float*)d_in[0];
    const int*   src   = (const int*)  d_in[1];
    const int*   dst   = (const int*)  d_in[2];
    const float* w     = (const float*)d_in[3];
    const float* Wm    = (const float*)d_in[4];
    const float* bias  = (const float*)d_in[5];
    const float* gamma = (const float*)d_in[6];
    const float* beta  = (const float*)d_in[7];
    float* out = (float*)d_out;

    const int N = in_sizes[0] / DFEAT;   // 50000
    const int E = in_sizes[1];           // 800000

    // workspace layout (assumes N % 16 == 0, as in reference)
    char* ws = (char*)d_ws;
    size_t off = 0;
    float* deg  = (float*)(ws + off); off += (size_t)N * sizeof(float);
    float* norm = (float*)(ws + off); off += (size_t)N * sizeof(float);
    unsigned short* Wb = (unsigned short*)(ws + off); off += (size_t)DFEAT * DFEAT * 2;
    off = (off + 255) & ~(size_t)255;
    float* agg = (float*)(ws + off); off += (size_t)N * DFEAT * sizeof(float);  // ~102.4 MB
    unsigned short* aggb = (unsigned short*)(ws + off);                          // ~51.2 MB

    zero_f4<<<1024, 256, 0, stream>>>((float4*)deg, (long)N / 4);
    zero_f4<<<4096, 256, 0, stream>>>((float4*)agg, (long)N * DFEAT / 4);

    deg_kernel <<<(E + 255) / 256, 256, 0, stream>>>(dst, deg, E);
    norm_kernel<<<(N + 255) / 256, 256, 0, stream>>>(deg, norm, N);
    bf16conv_kernel<<<512, 256, 0, stream>>>(
        (const float4*)Wm, (uint2*)Wb, (long)DFEAT * DFEAT / 4);
    scatter_kernel<<<(E + 7) / 8, 256, 0, stream>>>(feat, src, dst, w, norm, agg, E);
    bf16conv_kernel<<<4096, 256, 0, stream>>>(
        (const float4*)agg, (uint2*)aggb, (long)N * DFEAT / 4);
    gemm_ln_kernel<<<(N + 31) / 32, 256, 0, stream>>>(aggb, Wb, bias, gamma, beta, feat, out, N);
}
